// EnhancedAttention_8478265442475
// MI455X (gfx1250) — matrix operand order repaired and checked by reading, not compiled, not run on hardware
//
#include <hip/hip_runtime.h>
#include <hip/hip_bf16.h>

typedef __attribute__((ext_vector_type(16))) __bf16 v16bf;
typedef __attribute__((ext_vector_type(8)))  __bf16 v8bf;
typedef __attribute__((ext_vector_type(8)))  float  v8f;
typedef __attribute__((__vector_size__(4 * sizeof(int)))) int v4i;

#define WMMA_BF16(a, b, c) \
  __builtin_amdgcn_wmma_f32_16x16x32_bf16(false, (a), false, (b), (short)0, (c), false, false)

#if defined(__has_builtin)
#  if __has_builtin(__builtin_amdgcn_global_load_async_to_lds_b128)
#    define USE_ASYNC_LDS 1
#  else
#    define USE_ASYNC_LDS 0
#  endif
#else
#  define USE_ASYNC_LDS 0
#endif

static __device__ __forceinline__ v16bf join16(v8bf lo, v8bf hi) {
  union { v16bf v; v8bf h[2]; } u;
  u.h[0] = lo; u.h[1] = hi;
  return u.v;
}

#if USE_ASYNC_LDS
static __device__ __forceinline__ void async_cp16(const __bf16* g, __bf16* l) {
  __builtin_amdgcn_global_load_async_to_lds_b128(
      (__attribute__((address_space(1))) v4i*)(g),
      (__attribute__((address_space(3))) v4i*)(l), 0, 0);
}
#endif

// ---------------------------------------------------------------------------
// Projection GEMM: Y[4096x1024] = X[4096x1024] @ W[1024x1024] + b  (fp32 in)
// VTRANS=false: -> [B,H,S,Dh] bf16 (Q with scale folded, K)
// VTRANS=true : -> [B,H,Dh,S] bf16 (V, transposed for P@V B-fragments)
// 256 thr (8 waves), tile 128(M) x 64(N); double-buffered W tile in LDS.
// ---------------------------------------------------------------------------
template <bool VTRANS>
__global__ __launch_bounds__(256)
void proj_kernel(const float* __restrict__ X, const float* __restrict__ W,
                 const float* __restrict__ bias, __bf16* __restrict__ Out,
                 float scale) {
  __shared__ __bf16 wt[2][64][40];       // transposed W tiles [n][k], padded rows
  const int tid  = threadIdx.x;
  const int wave = tid >> 5;
  const int lane = tid & 31;
  const int kh   = lane >> 4;
  const int l16  = lane & 15;
  const int n0   = blockIdx.x * 64;
  const int m0   = blockIdx.y * 128;
  const int rowM = m0 + wave * 16 + l16;
  const int skk  = tid >> 6;             // staging id = tid + i*256 -> kk = skk + i*4
  const int snn  = tid & 63;

  v8f acc[4];
  #pragma unroll
  for (int j = 0; j < 4; ++j)
    #pragma unroll
    for (int r = 0; r < 8; ++r) acc[j][r] = 0.f;

  float wreg[8];
  #pragma unroll
  for (int i = 0; i < 8; ++i)            // stage tile 0
    wreg[i] = W[(size_t)(skk + i * 4) * 1024 + n0 + snn];
  #pragma unroll
  for (int i = 0; i < 8; ++i) wt[0][snn][skk + i * 4] = (__bf16)wreg[i];
  __syncthreads();

  for (int t = 0; t < 32; ++t) {
    const int k0 = t * 32;
    const int cur = t & 1, nxt = cur ^ 1;
    if (t < 31) {                        // next tile's global loads in flight
      #pragma unroll
      for (int i = 0; i < 8; ++i)
        wreg[i] = W[(size_t)(k0 + 32 + skk + i * 4) * 1024 + n0 + snn];
    }
    // A fragment (16x32): e<8 -> k = kh*8+e ; e>=8 -> k = 16+kh*8+(e-8)
    const float* xr = X + (size_t)rowM * 1024 + k0;
    float4 f0 = *(const float4*)(xr + kh * 8);
    float4 f1 = *(const float4*)(xr + kh * 8 + 4);
    float4 f2 = *(const float4*)(xr + 16 + kh * 8);
    float4 f3 = *(const float4*)(xr + 16 + kh * 8 + 4);
    v16bf a;
    a[0]  = (__bf16)f0.x; a[1]  = (__bf16)f0.y; a[2]  = (__bf16)f0.z; a[3]  = (__bf16)f0.w;
    a[4]  = (__bf16)f1.x; a[5]  = (__bf16)f1.y; a[6]  = (__bf16)f1.z; a[7]  = (__bf16)f1.w;
    a[8]  = (__bf16)f2.x; a[9]  = (__bf16)f2.y; a[10] = (__bf16)f2.z; a[11] = (__bf16)f2.w;
    a[12] = (__bf16)f3.x; a[13] = (__bf16)f3.y; a[14] = (__bf16)f3.z; a[15] = (__bf16)f3.w;

    #pragma unroll
    for (int j = 0; j < 4; ++j) {        // B fragment (32x16): e<8 -> k = kh*8+e ; e>=8 -> k = 16+kh*8+(e-8)
      v8bf b0 = *(const v8bf*)&wt[cur][j * 16 + l16][kh * 8];
      v8bf b1 = *(const v8bf*)&wt[cur][j * 16 + l16][16 + kh * 8];
      acc[j] = WMMA_BF16(a, join16(b0, b1), acc[j]);
    }
    if (t < 31) {
      #pragma unroll
      for (int i = 0; i < 8; ++i) wt[nxt][snn][skk + i * 4] = (__bf16)wreg[i];
    }
    __syncthreads();
  }

  // Epilogue: C layout -> VGPR r holds row m = r + kh*8, col = l16
  #pragma unroll
  for (int j = 0; j < 4; ++j) {
    int col = n0 + j * 16 + l16;
    float bb = bias[col];
    int h = col >> 6, d = col & 63;
    if (VTRANS) {
      int b = m0 >> 11;
      int s0r = (m0 & 2047) + wave * 16 + kh * 8;
      v8bf pk;
      #pragma unroll
      for (int r = 0; r < 8; ++r) pk[r] = (__bf16)((acc[j][r] + bb) * scale);
      *(v8bf*)&Out[(((size_t)b * 16 + h) * 64 + d) * 2048 + s0r] = pk;
    } else {
      #pragma unroll
      for (int r = 0; r < 8; ++r) {
        int row = m0 + wave * 16 + r + kh * 8;
        int b = row >> 11, s = row & 2047;
        Out[(((size_t)b * 16 + h) * 2048 + s) * 64 + d] =
            (__bf16)((acc[j][r] + bb) * scale);
      }
    }
  }
}

// ---------------------------------------------------------------------------
// Flash attention, block-shared K/V tiles.
// Block = 256 thr (8 waves) = 128 query rows of ONE (b,h); 64-key tiles of K
// and V staged in LDS (double buffered) and shared by all 8 waves -> 8x cut
// in L2 traffic. Per wave: 16 query rows x Dh=64, online softmax.
// Qb,Kb: [B,H,S,64] bf16 (scale folded into Q). Vt: [B,H,64,S] bf16.
// Ob: merged heads [B,S,1024] bf16.
// ---------------------------------------------------------------------------
__global__ __launch_bounds__(256)
void attn_kernel(const __bf16* __restrict__ Qb, const __bf16* __restrict__ Kb,
                 const __bf16* __restrict__ Vt, __bf16* __restrict__ Ob) {
  __shared__ __bf16 Kl[2][64][72];       // [s_local][d], padded rows (conflict-free)
  __shared__ __bf16 Vl[2][64][72];       // [d][s_local]
  __shared__ __bf16 P[8][16][72];        // per-wave P tile (C->A relayout)
  const int tid  = threadIdx.x;
  const int wave = tid >> 5;
  const int lane = tid & 31;
  const int kh   = lane >> 4;
  const int l16  = lane & 15;
  const int bh   = blockIdx.x >> 4;                    // 16 blocks per (b,h)
  const int q0   = (blockIdx.x & 15) * 128 + wave * 16;
  const int srow_ = tid >> 2;                          // staging row 0..63
  const int sseg  = (tid & 3) * 16;                    // staging col 0,16,32,48

  const __bf16* Kgb = Kb + ((size_t)bh * 2048 + srow_) * 64 + sseg; // + kb*64
  const __bf16* Vgb = Vt + ((size_t)bh * 64 + srow_) * 2048 + sseg; // + kb

  // Q A-fragments over d: runs [f*32+kh*8,+8) and [f*32+16+kh*8,+8)
  const __bf16* qr = Qb + ((size_t)bh * 2048 + q0 + l16) * 64;
  v16bf aq[2];
  #pragma unroll
  for (int f = 0; f < 2; ++f) {
    v8bf lo = *(const v8bf*)(qr + f * 32 + kh * 8);
    v8bf hi = *(const v8bf*)(qr + f * 32 + 16 + kh * 8);
    aq[f] = join16(lo, hi);
  }

  v8f co[4];
  float rmax[8], rsum[8];
  #pragma unroll
  for (int j = 0; j < 4; ++j)
    #pragma unroll
    for (int r = 0; r < 8; ++r) co[j][r] = 0.f;
  #pragma unroll
  for (int r = 0; r < 8; ++r) { rmax[r] = -3.0e38f; rsum[r] = 0.f; }

  // ---- stage tile 0 ----
#if USE_ASYNC_LDS
  async_cp16(Kgb,     &Kl[0][srow_][sseg]);
  async_cp16(Kgb + 8, &Kl[0][srow_][sseg + 8]);
  async_cp16(Vgb,     &Vl[0][srow_][sseg]);
  async_cp16(Vgb + 8, &Vl[0][srow_][sseg + 8]);
  asm volatile("s_wait_asynccnt 0x0" ::: "memory");
#else
  {
    v8bf k0 = *(const v8bf*)(Kgb), k1 = *(const v8bf*)(Kgb + 8);
    v8bf v0 = *(const v8bf*)(Vgb), v1 = *(const v8bf*)(Vgb + 8);
    *(v8bf*)&Kl[0][srow_][sseg]     = k0;
    *(v8bf*)&Kl[0][srow_][sseg + 8] = k1;
    *(v8bf*)&Vl[0][srow_][sseg]     = v0;
    *(v8bf*)&Vl[0][srow_][sseg + 8] = v1;
  }
#endif
  __syncthreads();

  for (int kb = 0; kb < 2048; kb += 64) {
    const int cur = (kb >> 6) & 1, nxt = cur ^ 1;
    const bool more = (kb + 64) < 2048;
#if USE_ASYNC_LDS
    if (more) {                          // async global->LDS for next tile
      const __bf16* kg = Kgb + (size_t)(kb + 64) * 64;
      const __bf16* vg = Vgb + (kb + 64);
      async_cp16(kg,     &Kl[nxt][srow_][sseg]);
      async_cp16(kg + 8, &Kl[nxt][srow_][sseg + 8]);
      async_cp16(vg,     &Vl[nxt][srow_][sseg]);
      async_cp16(vg + 8, &Vl[nxt][srow_][sseg + 8]);
    }
#else
    v8bf kr0, kr1, vr0, vr1;
    if (more) {                          // next tile's global loads in flight
      const __bf16* kg = Kgb + (size_t)(kb + 64) * 64;
      const __bf16* vg = Vgb + (kb + 64);
      kr0 = *(const v8bf*)(kg);
      kr1 = *(const v8bf*)(kg + 8);
      vr0 = *(const v8bf*)(vg);
      vr1 = *(const v8bf*)(vg + 8);
    }
#endif

    // ---- scores: four 16-key groups, K-dim = Dh = 64 (2 WMMAs each) ----
    v8f sc[4];
    #pragma unroll
    for (int g = 0; g < 4; ++g) {
      v8f z;
      #pragma unroll
      for (int r = 0; r < 8; ++r) z[r] = 0.f;
      #pragma unroll
      for (int f = 0; f < 2; ++f) {      // B frag: d runs [f*32+kh*8,+8) and [f*32+16+kh*8,+8)
        v8bf b0 = *(const v8bf*)&Kl[cur][g * 16 + l16][f * 32 + kh * 8];
        v8bf b1 = *(const v8bf*)&Kl[cur][g * 16 + l16][f * 32 + 16 + kh * 8];
        z = WMMA_BF16(aq[f], join16(b0, b1), z);
      }
      sc[g] = z;
    }

    // ---- preload V B-fragments (LDS) so they overlap the softmax VALU ----
    v16bf bvf[4][2];
    #pragma unroll
    for (int j = 0; j < 4; ++j)
      #pragma unroll
      for (int f = 0; f < 2; ++f) {
        v8bf b0 = *(const v8bf*)&Vl[cur][j * 16 + l16][f * 32 + kh * 8];
        v8bf b1 = *(const v8bf*)&Vl[cur][j * 16 + l16][f * 32 + 16 + kh * 8];
        bvf[j][f] = join16(b0, b1);
      }

    // ---- online softmax over 64 keys (row m = r + kh*8 across 16 lanes) ----
    #pragma unroll
    for (int r = 0; r < 8; ++r) {
      float t = fmaxf(fmaxf(sc[0][r], sc[1][r]), fmaxf(sc[2][r], sc[3][r]));
      #pragma unroll
      for (int off = 8; off; off >>= 1) t = fmaxf(t, __shfl_xor(t, off, 32));
      float nm = fmaxf(rmax[r], t);
      float corr = __expf(rmax[r] - nm);
      float p0 = __expf(sc[0][r] - nm);
      float p1 = __expf(sc[1][r] - nm);
      float p2 = __expf(sc[2][r] - nm);
      float p3 = __expf(sc[3][r] - nm);
      float ts = (p0 + p1) + (p2 + p3);
      #pragma unroll
      for (int off = 8; off; off >>= 1) ts += __shfl_xor(ts, off, 32);
      rsum[r] = rsum[r] * corr + ts;
      rmax[r] = nm;
      #pragma unroll
      for (int j = 0; j < 4; ++j) co[j][r] *= corr;
      int m = r + kh * 8;
      P[wave][m][l16]      = (__bf16)p0;
      P[wave][m][16 + l16] = (__bf16)p1;
      P[wave][m][32 + l16] = (__bf16)p2;
      P[wave][m][48 + l16] = (__bf16)p3;
    }
    asm volatile("" ::: "memory");       // keep DS write->read order (in-order per wave)

    // P as two A-fragments (16 rows x 64 keys)
    v16bf pa[2];
    #pragma unroll
    for (int f = 0; f < 2; ++f) {
      v8bf lo = *(const v8bf*)&P[wave][l16][f * 32 + kh * 8];
      v8bf hi = *(const v8bf*)&P[wave][l16][f * 32 + 16 + kh * 8];
      pa[f] = join16(lo, hi);
    }

    // ---- P @ V : 4 d-chunks x 2 key-halves ----
    #pragma unroll
    for (int j = 0; j < 4; ++j)
      #pragma unroll
      for (int f = 0; f < 2; ++f)
        co[j] = WMMA_BF16(pa[f], bvf[j][f], co[j]);

#if USE_ASYNC_LDS
    if (more) asm volatile("s_wait_asynccnt 0x0" ::: "memory");
#else
    if (more) {
      *(v8bf*)&Kl[nxt][srow_][sseg]     = kr0;
      *(v8bf*)&Kl[nxt][srow_][sseg + 8] = kr1;
      *(v8bf*)&Vl[nxt][srow_][sseg]     = vr0;
      *(v8bf*)&Vl[nxt][srow_][sseg + 8] = vr1;
    }
#endif
    __syncthreads();
  }

  // ---- normalize + store merged-head bf16 output ----
  const int bb = bh >> 4, hh = bh & 15;
  #pragma unroll
  for (int r = 0; r < 8; ++r) {
    float inv = 1.0f / rsum[r];
    int srow = (blockIdx.x & 15) * 128 + wave * 16 + r + kh * 8;
    #pragma unroll
    for (int j = 0; j < 4; ++j)
      Ob[((size_t)bb * 2048 + srow) * 1024 + hh * 64 + j * 16 + l16] =
          (__bf16)(co[j][r] * inv);
  }
}

// ---------------------------------------------------------------------------
// Output projection: out[4096x1024] = A(bf16)[4096x1024] @ Wo + bo  (fp32 out)
// ---------------------------------------------------------------------------
__global__ __launch_bounds__(256)
void outproj_kernel(const __bf16* __restrict__ A, const float* __restrict__ W,
                    const float* __restrict__ bias, float* __restrict__ Out) {
  __shared__ __bf16 wt[2][64][40];
  const int tid  = threadIdx.x;
  const int wave = tid >> 5;
  const int lane = tid & 31;
  const int kh   = lane >> 4;
  const int l16  = lane & 15;
  const int n0   = blockIdx.x * 64;
  const int m0   = blockIdx.y * 128;
  const int rowM = m0 + wave * 16 + l16;
  const int skk  = tid >> 6;
  const int snn  = tid & 63;

  v8f acc[4];
  #pragma unroll
  for (int j = 0; j < 4; ++j)
    #pragma unroll
    for (int r = 0; r < 8; ++r) acc[j][r] = 0.f;

  float wreg[8];
  #pragma unroll
  for (int i = 0; i < 8; ++i)
    wreg[i] = W[(size_t)(skk + i * 4) * 1024 + n0 + snn];
  #pragma unroll
  for (int i = 0; i < 8; ++i) wt[0][snn][skk + i * 4] = (__bf16)wreg[i];
  __syncthreads();

  for (int t = 0; t < 32; ++t) {
    const int k0 = t * 32;
    const int cur = t & 1, nxt = cur ^ 1;
    if (t < 31) {
      #pragma unroll
      for (int i = 0; i < 8; ++i)
        wreg[i] = W[(size_t)(k0 + 32 + skk + i * 4) * 1024 + n0 + snn];
    }
    const __bf16* ar = A + (size_t)rowM * 1024 + k0;
    v8bf a0 = *(const v8bf*)(ar + kh * 8);
    v8bf a1 = *(const v8bf*)(ar + 16 + kh * 8);
    v16bf a = join16(a0, a1);

    #pragma unroll
    for (int j = 0; j < 4; ++j) {
      v8bf b0 = *(const v8bf*)&wt[cur][j * 16 + l16][kh * 8];
      v8bf b1 = *(const v8bf*)&wt[cur][j * 16 + l16][16 + kh * 8];
      acc[j] = WMMA_BF16(a, join16(b0, b1), acc[j]);
    }
    if (t < 31) {
      #pragma unroll
      for (int i = 0; i < 8; ++i) wt[nxt][snn][skk + i * 4] = (__bf16)wreg[i];
    }
    __syncthreads();
  }

  #pragma unroll
  for (int j = 0; j < 4; ++j) {
    int col = n0 + j * 16 + l16;
    float bb = bias[col];
    #pragma unroll
    for (int r = 0; r < 8; ++r) {
      int row = m0 + wave * 16 + r + kh * 8;
      Out[(size_t)row * 1024 + col] = acc[j][r] + bb;
    }
  }
}

// ---------------------------------------------------------------------------
extern "C" void kernel_launch(void* const* d_in, const int* in_sizes, int n_in,
                              void* d_out, int out_size, void* d_ws, size_t ws_size,
                              hipStream_t stream) {
  const float* q  = (const float*)d_in[0];
  const float* k  = (const float*)d_in[1];
  const float* v  = (const float*)d_in[2];
  const float* Wq = (const float*)d_in[3];
  const float* bq = (const float*)d_in[4];
  const float* Wk = (const float*)d_in[5];
  const float* bk = (const float*)d_in[6];
  const float* Wv = (const float*)d_in[7];
  const float* bv = (const float*)d_in[8];
  const float* Wo = (const float*)d_in[9];
  const float* bo = (const float*)d_in[10];
  float* out = (float*)d_out;

  char* ws = (char*)d_ws;
  __bf16* Qb = (__bf16*)(ws);                          // [B,H,S,64]  8 MB
  __bf16* Kb = (__bf16*)(ws + ((size_t)8  << 20));     // [B,H,S,64]  8 MB
  __bf16* Vt = (__bf16*)(ws + ((size_t)16 << 20));     // [B,H,64,S]  8 MB
  __bf16* AO = (__bf16*)(ws + ((size_t)24 << 20));     // [B,S,1024]  8 MB

  dim3 gproj(16, 32);
  proj_kernel<false><<<gproj, 256, 0, stream>>>(q, Wq, bq, Qb, 0.03125f); // 1/sqrt(1024)
  proj_kernel<false><<<gproj, 256, 0, stream>>>(k, Wk, bk, Kb, 1.0f);
  proj_kernel<true ><<<gproj, 256, 0, stream>>>(v, Wv, bv, Vt, 1.0f);
  attn_kernel<<<512, 256, 0, stream>>>(Qb, Kb, Vt, AO);
  outproj_kernel<<<gproj, 256, 0, stream>>>(AO, Wo, bo, out);
}
